// CausalSelfAttention_86560770883928
// MI455X (gfx1250) — matrix operand order repaired and checked by reading, not compiled, not run on hardware
//
#include <hip/hip_runtime.h>
#include <hip/hip_bf16.h>

typedef __bf16 bf16;
typedef __attribute__((ext_vector_type(16))) __bf16 v16bf;
typedef __attribute__((ext_vector_type(8)))  float  v8f;

#define DEV __device__ __forceinline__

constexpr int   Bb = 2, Tt = 2048, Cc = 1024, Hh = 16, Dd = 64;
constexpr float RMS_EPS  = 1.1920928955078125e-07f;
constexpr float NEG_BIG  = -3.0e38f;
constexpr float LN_10000 = 9.210340371976184f;

// Padded LDS row strides (all 16B multiples; chosen so row*stride mod 64 banks
// is spread: 72 bf16 = 36 dwords -> distinct banks for rows 0..15).
constexpr int STR_G = 72;   // GEMM staging rows (64 data + 8 pad)
constexpr int STR_V = 40;   // V-transposed rows (32 data + 8 pad)
constexpr int STR_P = 40;   // P staging rows   (32 data + 8 pad)

union FragBF { v16bf v; uint4 q[2]; };

// ---- WMMA fragment loaders (ISA 7.12.2 layouts, wave32) ----
// A-fragment 16x32 bf16 (MxK), source row-major [m][k], row stride ld.
// lanes 0-15: row=lane, K={0..7,16..23}; lanes 16-31: same row, K={8..15,24..31}
DEV v16bf frag_a(const bf16* p, int ld, int lane) {
  int row = lane & 15;
  int ko  = (lane >> 4) << 3;
  FragBF f;
  f.q[0] = *(const uint4*)(p + row * ld + ko);
  f.q[1] = *(const uint4*)(p + row * ld + ko + 16);
  return f.v;
}
// B-fragment 32x16 bf16 (KxN) from source stored row-major [n][k] (i.e. B^T).
// lanes 0-15: col=lane, K={0..7,16..23}; lanes 16-31: col=lane-16, K={8..15,24..31}
DEV v16bf frag_b_nk(const bf16* p, int ld, int lane) {
  int n  = lane & 15;
  int ko = (lane >> 4) << 3;
  FragBF f;
  f.q[0] = *(const uint4*)(p + n * ld + ko);
  f.q[1] = *(const uint4*)(p + n * ld + ko + 16);
  return f.v;
}

DEV v8f wmma_bf16(v16bf a, v16bf b, v8f c) {
  return __builtin_amdgcn_wmma_f32_16x16x32_bf16(false, a, false, b,
                                                 (short)0, c, false, false);
}

DEV void cvt8_store(bf16* dst, float4 a, float4 b) {
  union { bf16 h[8]; uint4 u; } pk;
  pk.h[0]=(bf16)a.x; pk.h[1]=(bf16)a.y; pk.h[2]=(bf16)a.z; pk.h[3]=(bf16)a.w;
  pk.h[4]=(bf16)b.x; pk.h[5]=(bf16)b.y; pk.h[6]=(bf16)b.z; pk.h[7]=(bf16)b.w;
  *(uint4*)dst = pk.u;
}

// Async global->LDS 16-byte copy (ASYNCcnt-tracked, ISA §10.7 / 08_async_tensor).
// ldsoff = low 32 bits of the generic pointer == LDS byte offset (aperture rule).
DEV void async_g2l_b128(unsigned ldsoff, const bf16* g) {
  asm volatile("global_load_async_to_lds_b128 %0, %1, off"
               :: "v"(ldsoff), "v"(g) : "memory");
}
DEV void wait_async0() { asm volatile("s_wait_asynccnt 0" ::: "memory"); }

// ============================================================================
// Kernel 1 (fused): q,k,v = x @ w_{q,k,v}^T for one 64-row x one-head tile.
// x tile staged once per K-step, 12 WMMAs per barrier pair. All accumulators
// spilled to LDS before the RMS+RoPE epilogue (no scratch spills).
// Outputs bf16 [B][H][T][D].
// ============================================================================
__global__ __launch_bounds__(256) void qkv_rope_kernel(
    const float* __restrict__ x,   const float* __restrict__ w_q,
    const float* __restrict__ w_k, const float* __restrict__ w_v,
    bf16* __restrict__ qr, bf16* __restrict__ kr, bf16* __restrict__ vr) {
  // staging: 4 * 64*72*2 = 36864 B ; epilogue: 3 * 64*68*4 = 52224 B
  __shared__ __align__(16) char smem[3 * 64 * 68 * 4];
  bf16*  lx   = (bf16*)smem;                    // [64][STR_G]
  float* lacc = (float*)smem;                   // 3 planes [64][68]

  const int tid = threadIdx.x, lane = tid & 31, wav = tid >> 5;
  const int h   = blockIdx.x;                   // head == 64-col tile
  const int rg0 = blockIdx.y * 64;              // row base (b*T + t)
  const int mi  = wav >> 1, ni0 = (wav & 1) * 2;

  const float* wsrc[3] = { w_q, w_k, w_v };
  bf16*        dsts[3] = { qr,  kr,  vr  };
  bf16*        lw[3]   = { lx + 64 * STR_G, lx + 2 * 64 * STR_G,
                           lx + 3 * 64 * STR_G };

  v8f acc[3][2] = {};

  const int sr = tid >> 2;                      // staging row 0..63
  const int sk = (tid & 3) * 16;                // staging k base

  for (int k0 = 0; k0 < Cc; k0 += 64) {
    {
      const float4* gx = (const float4*)(x + (size_t)(rg0 + sr) * Cc + k0 + sk);
      float4 a0 = gx[0], a1 = gx[1], a2 = gx[2], a3 = gx[3];
      cvt8_store(lx + sr * STR_G + sk,     a0, a1);
      cvt8_store(lx + sr * STR_G + sk + 8, a2, a3);
    }
#pragma unroll
    for (int z = 0; z < 3; ++z) {
      const float4* gw =
          (const float4*)(wsrc[z] + (size_t)(h * 64 + sr) * Cc + k0 + sk);
      float4 b0 = gw[0], b1 = gw[1], b2 = gw[2], b3 = gw[3];
      cvt8_store(lw[z] + sr * STR_G + sk,     b0, b1);
      cvt8_store(lw[z] + sr * STR_G + sk + 8, b2, b3);
    }
    __syncthreads();
#pragma unroll
    for (int kk = 0; kk < 64; kk += 32) {
      v16bf af = frag_a(lx + mi * 16 * STR_G + kk, STR_G, lane);
      v16bf bz[6];
#pragma unroll
      for (int z = 0; z < 3; ++z) {
        bz[2*z]   = frag_b_nk(lw[z] + (ni0    ) * 16 * STR_G + kk, STR_G, lane);
        bz[2*z+1] = frag_b_nk(lw[z] + (ni0 + 1) * 16 * STR_G + kk, STR_G, lane);
      }
#pragma unroll
      for (int z = 0; z < 3; ++z) {
        acc[z][0] = wmma_bf16(af, bz[2*z],   acc[z][0]);
        acc[z][1] = wmma_bf16(af, bz[2*z+1], acc[z][1]);
      }
    }
    __syncthreads();
  }

  // spill all three planes, releasing accumulator registers before epilogue
  const int half = lane >> 4, nl = lane & 15;
#pragma unroll
  for (int z = 0; z < 3; ++z) {
    float* pl = lacc + z * 64 * 68;
#pragma unroll
    for (int r = 0; r < 8; ++r) {
      int m = mi * 16 + r + half * 8;
      pl[m * 68 + (ni0    ) * 16 + nl] = acc[z][0][r];
      pl[m * 68 + (ni0 + 1) * 16 + nl] = acc[z][1][r];
    }
  }
  __syncthreads();

  const int b = rg0 / Tt, t0 = rg0 % Tt;
#pragma unroll 1
  for (int z = 0; z < 2; ++z) {                 // q,k: RMS + RoPE
    if (tid < 64) {
      const float* row = lacc + z * 64 * 68 + tid * 68;
      float ss = 0.f;
#pragma unroll
      for (int i = 0; i < 64; ++i) { float v = row[i]; ss += v * v; }
      float rms = rsqrtf(ss * (1.0f / 64.0f) + RMS_EPS);
      int t = t0 + tid;
      size_t ob = (((size_t)b * Hh + h) * Tt + t) * Dd;
      bf16* dst = dsts[z];
#pragma unroll 4
      for (int i = 0; i < 32; ++i) {
        float x1 = row[i] * rms, x2 = row[i + 32] * rms;
        float inv = __expf(-LN_10000 * (float)(2 * i) * (1.0f / 64.0f));
        float ang = (float)t * inv;
        float sn, cs; __sincosf(ang, &sn, &cs);
        dst[ob + i]      = (bf16)( x1 * cs + x2 * sn);
        dst[ob + i + 32] = (bf16)(-x1 * sn + x2 * cs);
      }
    }
  }
  {                                             // v: passthrough
    const float* pl = lacc + 2 * 64 * 68;
    for (int e = tid; e < 64 * 64; e += 256) {
      int m = e >> 6, d = e & 63;
      dsts[2][(((size_t)b * Hh + h) * Tt + (t0 + m)) * Dd + d] =
          (bf16)pl[m * 68 + d];
    }
  }
}

// ============================================================================
// Kernel 2: flash attention. One (b,h) per blockIdx.y, 128 queries per block,
// one 16-query tile per wave. K staged via async global->LDS; V staged
// transposed; padded LDS strides for conflict-free b128 fragment loads.
// ============================================================================
__global__ __launch_bounds__(256) void flash_attn_kernel(
    const bf16* __restrict__ qr, const bf16* __restrict__ kr,
    const bf16* __restrict__ vr, bf16* __restrict__ o) {
  __shared__ __align__(16) bf16 lK [32 * STR_G];     // [key][d]
  __shared__ __align__(16) bf16 lVt[64 * STR_V];     // [d][key]
  __shared__ __align__(16) bf16 lP [8][16 * STR_P];  // per-wave P staging

  const int tid = threadIdx.x, lane = tid & 31, wav = tid >> 5;
  const int q0  = blockIdx.x * 128;
  const int bh  = blockIdx.y;
  const size_t base = (size_t)bh * Tt * Dd;

  const int qrow = q0 + wav * 16;
  v16bf aq0 = frag_a(qr + base + (size_t)qrow * Dd,      Dd, lane);
  v16bf aq1 = frag_a(qr + base + (size_t)qrow * Dd + 32, Dd, lane);

  v8f   oa[4] = {};
  float mrow[8], lrow[8];
#pragma unroll
  for (int r = 0; r < 8; ++r) { mrow[r] = NEG_BIG; lrow[r] = 0.f; }

  const int half = lane >> 4, nl = lane & 15;
  const int nch  = blockIdx.x * 4 + 4;               // causal chunk bound
  const int lm   = tid >> 3, ld8 = (tid & 7) * 8;    // staging coords

  const unsigned ldsK = (unsigned)(size_t)(lK + lm * STR_G + ld8);

  for (int c = 0; c < nch; ++c) {
    __syncthreads();
    // K chunk: async DMA straight into LDS (ASYNCcnt)
    async_g2l_b128(ldsK, kr + base + (size_t)(c * 32 + lm) * Dd + ld8);
    // V chunk: load + transpose-scatter into [d][key]
    {
      union { uint4 u; bf16 h[8]; } vu;
      vu.u = *(const uint4*)(vr + base + (size_t)(c * 32 + lm) * Dd + ld8);
#pragma unroll
      for (int j = 0; j < 8; ++j) lVt[(ld8 + j) * STR_V + lm] = vu.h[j];
    }
    wait_async0();
    __syncthreads();

    if (c * 32 <= qrow + 15) {  // wave-uniform skip of fully-masked chunks
      // S = Q K^T : preload all K fragments, then WMMA burst
      v16bf bk[4];
      bk[0] = frag_b_nk(lK,                    STR_G, lane);
      bk[1] = frag_b_nk(lK + 32,               STR_G, lane);
      bk[2] = frag_b_nk(lK + 16 * STR_G,       STR_G, lane);
      bk[3] = frag_b_nk(lK + 16 * STR_G + 32,  STR_G, lane);
      v8f s0 = {}, s1 = {};
      s0 = wmma_bf16(aq0, bk[0], s0);
      s0 = wmma_bf16(aq1, bk[1], s0);
      s1 = wmma_bf16(aq0, bk[2], s1);
      s1 = wmma_bf16(aq1, bk[3], s1);

      const float scale = 0.125f;  // 1/sqrt(64)
#pragma unroll
      for (int r = 0; r < 8; ++r) {
        int qg  = qrow + r + half * 8;
        int kg0 = c * 32 + nl;
        float v0 = (kg0      <= qg) ? s0[r] * scale : NEG_BIG;
        float v1 = (kg0 + 16 <= qg) ? s1[r] * scale : NEG_BIG;
        float mx = fmaxf(v0, v1);
        mx = fmaxf(mx, __shfl_xor(mx, 1, 16));
        mx = fmaxf(mx, __shfl_xor(mx, 2, 16));
        mx = fmaxf(mx, __shfl_xor(mx, 4, 16));
        mx = fmaxf(mx, __shfl_xor(mx, 8, 16));
        float mnew  = fmaxf(mrow[r], mx);
        float alpha = __expf(mrow[r] - mnew);
        float e0 = __expf(v0 - mnew);
        float e1 = __expf(v1 - mnew);
        float rs = e0 + e1;
        rs += __shfl_xor(rs, 1, 16);
        rs += __shfl_xor(rs, 2, 16);
        rs += __shfl_xor(rs, 4, 16);
        rs += __shfl_xor(rs, 8, 16);
        lrow[r] = lrow[r] * alpha + rs;
        mrow[r] = mnew;
#pragma unroll
        for (int sl = 0; sl < 4; ++sl) oa[sl][r] *= alpha;
        lP[wav][(r + half * 8) * STR_P + nl]      = (bf16)e0;
        lP[wav][(r + half * 8) * STR_P + nl + 16] = (bf16)e1;
      }
      // V fragments are independent of the P staging writes
      v16bf bv[4];
#pragma unroll
      for (int sl = 0; sl < 4; ++sl)
        bv[sl] = frag_b_nk(lVt + sl * 16 * STR_V, STR_V, lane);
      // wave-internal LDS round-trip: P (D layout) -> A-fragment layout
      asm volatile("s_wait_dscnt 0" ::: "memory");
      __builtin_amdgcn_wave_barrier();
      v16bf ap = frag_a(lP[wav], STR_P, lane);
#pragma unroll
      for (int sl = 0; sl < 4; ++sl)
        oa[sl] = wmma_bf16(ap, bv[sl], oa[sl]);
    }
  }

  // O as bf16 [B][T][H][D] (== [B][T][C]) for the projection GEMM
  const int b = bh / Hh, h = bh % Hh;
#pragma unroll
  for (int sl = 0; sl < 4; ++sl) {
#pragma unroll
    for (int r = 0; r < 8; ++r) {
      int t = qrow + r + half * 8;
      int d = sl * 16 + nl;
      o[(((size_t)b * Tt + t) * Hh + h) * Dd + d] = (bf16)(oa[sl][r] / lrow[r]);
    }
  }
}

// ============================================================================
// Kernel 3: out = O @ w_proj^T, f32 output. 64x64 tile, K=64 staging,
// bf16 A-tile staged via async global->LDS.
// ============================================================================
__global__ __launch_bounds__(256) void proj_kernel(
    const bf16* __restrict__ oin, const float* __restrict__ wp,
    float* __restrict__ out) {
  __shared__ __align__(16) bf16 lx[64 * STR_G];
  __shared__ __align__(16) bf16 lw[64 * STR_G];
  const int tid = threadIdx.x, lane = tid & 31, wav = tid >> 5;
  const int cg0 = blockIdx.x * 64, rg0 = blockIdx.y * 64;
  const int mi = wav >> 1, ni0 = (wav & 1) * 2;
  v8f acc0 = {}, acc1 = {};
  const int sr = tid >> 2, sk = (tid & 3) * 16;

  const unsigned lxoff = (unsigned)(size_t)(lx + sr * STR_G + sk);

  for (int k0 = 0; k0 < Cc; k0 += 64) {
    {
      const bf16* go = oin + (size_t)(rg0 + sr) * Cc + k0 + sk;
      async_g2l_b128(lxoff,      go);
      async_g2l_b128(lxoff + 16, go + 8);
    }
    {
      const float4* gw = (const float4*)(wp + (size_t)(cg0 + sr) * Cc + k0 + sk);
      float4 b0 = gw[0], b1 = gw[1], b2 = gw[2], b3 = gw[3];
      cvt8_store(lw + sr * STR_G + sk,     b0, b1);
      cvt8_store(lw + sr * STR_G + sk + 8, b2, b3);
    }
    wait_async0();
    __syncthreads();
#pragma unroll
    for (int kk = 0; kk < 64; kk += 32) {
      v16bf af = frag_a(lx + mi * 16 * STR_G + kk, STR_G, lane);
      v16bf b0 = frag_b_nk(lw + (ni0    ) * 16 * STR_G + kk, STR_G, lane);
      v16bf b1 = frag_b_nk(lw + (ni0 + 1) * 16 * STR_G + kk, STR_G, lane);
      acc0 = wmma_bf16(af, b0, acc0);
      acc1 = wmma_bf16(af, b1, acc1);
    }
    __syncthreads();
  }
  const int half = lane >> 4, nl = lane & 15;
#pragma unroll
  for (int r = 0; r < 8; ++r) {
    int m = rg0 + mi * 16 + r + half * 8;
    out[(size_t)m * Cc + cg0 + (ni0    ) * 16 + nl] = acc0[r];
    out[(size_t)m * Cc + cg0 + (ni0 + 1) * 16 + nl] = acc1[r];
  }
}

// ============================================================================
extern "C" void kernel_launch(void* const* d_in, const int* in_sizes, int n_in,
                              void* d_out, int out_size, void* d_ws, size_t ws_size,
                              hipStream_t stream) {
  (void)in_sizes; (void)n_in; (void)out_size; (void)ws_size;
  const float* x   = (const float*)d_in[0];
  const float* w_q = (const float*)d_in[1];
  const float* w_k = (const float*)d_in[2];
  const float* w_v = (const float*)d_in[3];
  const float* w_p = (const float*)d_in[4];
  float* out = (float*)d_out;

  const size_t per = (size_t)Bb * Hh * Tt * Dd;   // 4,194,304 elems (8 MB bf16)
  bf16* qr = (bf16*)d_ws;
  bf16* kr = qr + per;
  bf16* vr = kr + per;
  bf16* ob = vr + per;

  qkv_rope_kernel<<<dim3(Hh, Bb * Tt / 64), 256, 0, stream>>>(
      x, w_q, w_k, w_v, qr, kr, vr);
  flash_attn_kernel<<<dim3(Tt / 128, Bb * Hh), 256, 0, stream>>>(
      qr, kr, vr, ob);
  proj_kernel<<<dim3(Cc / 64, Bb * Tt / 64), 256, 0, stream>>>(
      ob, w_p, out);
}
